// SampledARHead_51616916963558
// MI455X (gfx1250) — compile-verified, matrix operand order checked by reading, not run on hardware
//
#include <hip/hip_runtime.h>
#include <math.h>

typedef _Float16 f16;
typedef __attribute__((ext_vector_type(4)))  _Float16 v4h;
typedef __attribute__((ext_vector_type(8)))  _Float16 v8h;
typedef __attribute__((ext_vector_type(16))) _Float16 v16h;
typedef __attribute__((ext_vector_type(8)))  float    v8f;

#define VOCAB    1000000
#define DIM      128
#define N_ROWS   16384
#define NSAMP    8192
#define INV_TEMP 20.0f
#define LOG2E    1.44269504088896340736f
#define LN2      0.69314718055994530942f

__device__ __forceinline__ int clampi(int v, int lo, int hi) {
    return v < lo ? lo : (v > hi ? hi : v);
}

// ---------------------------------------------------------------------------
// Kernel A: gather negative embeddings, L2-normalize, emit f16 [NSAMP][128].
// One wave per sample; lane handles 4 consecutive floats (coalesced float4).
// ---------------------------------------------------------------------------
__global__ __launch_bounds__(256) void neg_norm_kernel(
    const float* __restrict__ proj, const int* __restrict__ neg_ids,
    f16* __restrict__ neg16)
{
    const int wid  = (blockIdx.x * blockDim.x + threadIdx.x) >> 5;
    const int lane = threadIdx.x & 31;
    if (wid >= NSAMP) return;

    const int id = clampi(neg_ids[wid], 0, VOCAB - 1);
    const float4 v = ((const float4*)(proj + (size_t)id * DIM))[lane];

    float pp = v.x * v.x + v.y * v.y + v.z * v.z + v.w * v.w;
    #pragma unroll
    for (int m = 16; m >= 1; m >>= 1) pp += __shfl_xor(pp, m);

    const float inv = 1.0f / (sqrtf(pp) + 1e-12f);
    v4h h;
    h.x = (f16)(v.x * inv); h.y = (f16)(v.y * inv);
    h.z = (f16)(v.z * inv); h.w = (f16)(v.w * inv);
    ((v4h*)(neg16 + (size_t)wid * DIM))[lane] = h;
}

// ---------------------------------------------------------------------------
// Kernel B: per input row — L2-normalize x (emit f16), gather+normalize the
// positive embedding, compute pos_logit = cos(x,p)/TEMP and effective weight.
// One wave per row.
// ---------------------------------------------------------------------------
__global__ __launch_bounds__(256) void row_kernel(
    const float* __restrict__ inputs, const int* __restrict__ target_ids,
    const float* __restrict__ weights, const float* __restrict__ proj,
    f16* __restrict__ x16, float* __restrict__ pos_logit,
    float* __restrict__ w_eff)
{
    const int wid  = (blockIdx.x * blockDim.x + threadIdx.x) >> 5;
    const int lane = threadIdx.x & 31;
    if (wid >= N_ROWS) return;

    const int t   = target_ids[wid];          // [N,1] flattened
    const int tgt = clampi(t, 0, VOCAB - 1);

    const float4 x = ((const float4*)(inputs + (size_t)wid * DIM))[lane];
    const float4 p = ((const float4*)(proj   + (size_t)tgt * DIM))[lane];

    float xx = x.x * x.x + x.y * x.y + x.z * x.z + x.w * x.w;
    float pp = p.x * p.x + p.y * p.y + p.z * p.z + p.w * p.w;
    float xp = x.x * p.x + x.y * p.y + x.z * p.z + x.w * p.w;
    #pragma unroll
    for (int m = 16; m >= 1; m >>= 1) {
        xx += __shfl_xor(xx, m);
        pp += __shfl_xor(pp, m);
        xp += __shfl_xor(xp, m);
    }
    const float ix = 1.0f / (sqrtf(xx) + 1e-12f);
    const float ip = 1.0f / (sqrtf(pp) + 1e-12f);

    v4h h;
    h.x = (f16)(x.x * ix); h.y = (f16)(x.y * ix);
    h.z = (f16)(x.z * ix); h.w = (f16)(x.w * ix);
    ((v4h*)(x16 + (size_t)wid * DIM))[lane] = h;

    if (lane == 0) {
        pos_logit[wid] = xp * ix * ip * INV_TEMP;
        w_eff[wid]     = weights[wid] * ((t != -100) ? 1.0f : 0.0f);
    }
}

// ---------------------------------------------------------------------------
// Kernel C: fused GEMM + streaming sum-of-exp.
// One wave owns TWO 16-row A tiles (32 rows, 64 VGPRs, register-resident) and
// sweeps HALF the column space (4096 cols -> 256 tiles), so each B fragment
// feeds two chained v_wmma_f32_16x16x32_f16 -> B traffic halved vs 1 tile/wave
// while keeping 1024 waves in flight. The 16x16 f32 results fold into per-row
// sum-of-exp accumulators (logits bounded +/-20 => no max subtraction).
// Column halves write disjoint partial arrays (deterministic, no atomics).
//
// 16-bit operand layout (ISA 7.12.2): lane<16 holds row=lane, K {0..7,16..23};
// lane>=16 holds row=lane-16, K {8..15,24..31} -> two b128 loads per chunk.
// C layout: VGPR v, lanes 0-15: (M=v, N=lane); lanes 16-31: (M=8+v, N=lane-16).
// ---------------------------------------------------------------------------
__global__ __launch_bounds__(256) void gemm_lse_kernel(
    const f16* __restrict__ x16, const f16* __restrict__ neg16,
    float* __restrict__ sum_exp0, float* __restrict__ sum_exp1)
{
    const int wid  = (blockIdx.x * blockDim.x + threadIdx.x) >> 5; // [0,1024)
    const int lane = threadIdx.x & 31;

    const int row_block = wid >> 1;          // [0,512): 32-row block
    const int col_half  = wid & 1;           // [0,2): 4096-column half
    const int R         = row_block * 32;
    const int r         = lane & 15;
    const int koff      = (lane >> 4) * 8;

    // Load & keep two A tiles: 2 x 4 K-chunks x 16 halves per lane (64 VGPRs).
    v16h a[2][4];
    #pragma unroll
    for (int t = 0; t < 2; ++t) {
        const f16* arow = x16 + (size_t)(R + t * 16 + r) * DIM;
        #pragma unroll
        for (int kc = 0; kc < 4; ++kc) {
            const int K0 = kc * 32 + koff;
            v8h lo = *(const v8h*)(arow + K0);
            v8h hi = *(const v8h*)(arow + K0 + 16);
            a[t][kc] = __builtin_shufflevector(lo, hi, 0, 1, 2, 3, 4, 5, 6, 7,
                                                       8, 9, 10, 11, 12, 13, 14, 15);
        }
    }

    float acc0[8] = {0.f, 0.f, 0.f, 0.f, 0.f, 0.f, 0.f, 0.f};
    float acc1[8] = {0.f, 0.f, 0.f, 0.f, 0.f, 0.f, 0.f, 0.f};

    const f16* bbase = neg16 + (size_t)col_half * (NSAMP / 2) * DIM;

    for (int ct = 0; ct < (NSAMP / 2) / 16; ++ct) {
        const f16* bcol = bbase + (size_t)(ct * 16 + r) * DIM;
        v8f c0 = {};
        v8f c1 = {};
        #pragma unroll
        for (int kc = 0; kc < 4; ++kc) {
            const int K0 = kc * 32 + koff;
            v8h lo = *(const v8h*)(bcol + K0);
            v8h hi = *(const v8h*)(bcol + K0 + 16);
            v16h b = __builtin_shufflevector(lo, hi, 0, 1, 2, 3, 4, 5, 6, 7,
                                                     8, 9, 10, 11, 12, 13, 14, 15);
            c0 = __builtin_amdgcn_wmma_f32_16x16x32_f16(
                     false, a[0][kc], false, b, (short)0, c0, false, false);
            c1 = __builtin_amdgcn_wmma_f32_16x16x32_f16(
                     false, a[1][kc], false, b, (short)0, c1, false, false);
        }
        // exp(cos/TEMP) = exp2(cos * INV_TEMP * log2(e)); bounded, no max needed.
        #pragma unroll
        for (int i = 0; i < 8; ++i) {
            acc0[i] += __builtin_amdgcn_exp2f(c0[i] * (INV_TEMP * LOG2E));
            acc1[i] += __builtin_amdgcn_exp2f(c1[i] * (INV_TEMP * LOG2E));
        }
    }

    // Row sums: reduce across each 16-lane half (xor stays within the half).
    #pragma unroll
    for (int i = 0; i < 8; ++i) {
        float s0 = acc0[i];
        float s1 = acc1[i];
        s0 += __shfl_xor(s0, 1); s1 += __shfl_xor(s1, 1);
        s0 += __shfl_xor(s0, 2); s1 += __shfl_xor(s1, 2);
        s0 += __shfl_xor(s0, 4); s1 += __shfl_xor(s1, 4);
        s0 += __shfl_xor(s0, 8); s1 += __shfl_xor(s1, 8);
        acc0[i] = s0;
        acc1[i] = s1;
    }
    float* out = col_half ? sum_exp1 : sum_exp0;
    if ((lane & 15) == 0) {
        const int mbase = R + (lane >> 4) * 8;
        #pragma unroll
        for (int i = 0; i < 8; ++i) {
            out[mbase + i]      = acc0[i];
            out[mbase + 16 + i] = acc1[i];
        }
    }
}

// ---------------------------------------------------------------------------
// Kernel D: ce[n] = log(se0[n] + se1[n] + exp(pos_logit[n])) - pos_logit[n];
// loss = sum(ce*w) / max(sum(w), 1e-12). Single-block reduction.
// ---------------------------------------------------------------------------
__global__ __launch_bounds__(256) void loss_kernel(
    const float* __restrict__ sum_exp0, const float* __restrict__ sum_exp1,
    const float* __restrict__ pos_logit, const float* __restrict__ w_eff,
    float* __restrict__ out)
{
    __shared__ float sn[256];
    __shared__ float sd[256];

    float num = 0.f, den = 0.f;
    for (int n = threadIdx.x; n < N_ROWS; n += blockDim.x) {
        const float pl    = pos_logit[n];
        const float total = sum_exp0[n] + sum_exp1[n] +
                            __builtin_amdgcn_exp2f(pl * LOG2E);
        const float ce    = __builtin_amdgcn_logf(total) * LN2 - pl;
        const float w     = w_eff[n];
        num += ce * w;
        den += w;
    }
    sn[threadIdx.x] = num;
    sd[threadIdx.x] = den;
    __syncthreads();
    for (int s = 128; s > 0; s >>= 1) {
        if ((int)threadIdx.x < s) {
            sn[threadIdx.x] += sn[threadIdx.x + s];
            sd[threadIdx.x] += sd[threadIdx.x + s];
        }
        __syncthreads();
    }
    if (threadIdx.x == 0) out[0] = sn[0] / fmaxf(sd[0], 1e-12f);
}

// ---------------------------------------------------------------------------
extern "C" void kernel_launch(void* const* d_in, const int* in_sizes, int n_in,
                              void* d_out, int out_size, void* d_ws, size_t ws_size,
                              hipStream_t stream)
{
    const float* inputs     = (const float*)d_in[0];  // [N,128] f32
    const int*   target_ids = (const int*)  d_in[1];  // [N,1]   i32
    const int*   neg_ids    = (const int*)  d_in[2];  // [S]     i32
    const float* weights    = (const float*)d_in[3];  // [N]     f32
    const float* proj       = (const float*)d_in[4];  // [V,128] f32

    char* ws = (char*)d_ws;
    f16*   x16       = (f16*)(ws);                                   // 4 MiB
    f16*   neg16     = (f16*)(ws + (size_t)N_ROWS * DIM * 2);        // 2 MiB
    float* pos_logit = (float*)(ws + (size_t)(N_ROWS + NSAMP) * DIM * 2);
    float* w_eff     = pos_logit + N_ROWS;
    float* sum_exp0  = w_eff + N_ROWS;
    float* sum_exp1  = sum_exp0 + N_ROWS;

    neg_norm_kernel<<<NSAMP / 8, 256, 0, stream>>>(proj, neg_ids, neg16);
    row_kernel<<<N_ROWS / 8, 256, 0, stream>>>(inputs, target_ids, weights,
                                               proj, x16, pos_logit, w_eff);
    gemm_lse_kernel<<<1024 / 8, 256, 0, stream>>>(x16, neg16, sum_exp0, sum_exp1);
    loss_kernel<<<1, 256, 0, stream>>>(sum_exp0, sum_exp1, pos_logit, w_eff,
                                       (float*)d_out);
}
